// AGNLayer_21792664060333
// MI455X (gfx1250) — matrix operand order repaired and checked by reading, not compiled, not run on hardware
//
#include <hip/hip_runtime.h>
#include <cfloat>

#define NN 768
#define HH 128
constexpr int NROWS = NN * NN;              // 589824
constexpr int ROWS_PER_BLK = 64;
constexpr int GEMM_BLOCKS = NROWS / ROWS_PER_BLK;   // 9216

typedef _Float16 v16h  __attribute__((ext_vector_type(16)));
typedef float    v8f   __attribute__((ext_vector_type(8)));

// ---------------- small linear: Y = X @ W^T + b  ([768,128]x[128,128]) ----
__global__ void linear_kernel(const float* __restrict__ X, const float* __restrict__ W,
                              const float* __restrict__ b, float* __restrict__ Y) {
    int idx = blockIdx.x * 256 + threadIdx.x;     // 98304 outputs
    int i = idx >> 7, g = idx & 127;
    const float* xr = X + (size_t)i * HH;
    const float* wr = W + (size_t)g * HH;
    float acc = 0.f;
#pragma unroll 8
    for (int h = 0; h < HH; ++h) acc = fmaf(xr[h], wr[h], acc);
    Y[idx] = acc + b[g];
}

// ---------------- eU_w -> f16 ---------------------------------------------
__global__ void cvt_f16_kernel(const float* __restrict__ W, _Float16* __restrict__ Wh) {
    int idx = blockIdx.x * 256 + threadIdx.x;     // 16384
    Wh[idx] = (_Float16)W[idx];
}

// ---------------- main edge GEMM + gate + BN partials (WMMA) --------------
// 128 threads = 4 waves; each wave computes 16 rows x 128 channels.
// e rows are staged global->LDS with CDNA5 async-to-LDS (ASYNCcnt) ops.
__global__ __launch_bounds__(128) void edge_gemm_kernel(
    const float* __restrict__ e, const _Float16* __restrict__ Wh,
    const float* __restrict__ bias, const float* __restrict__ Vx,
    float* __restrict__ enew, float* __restrict__ psum, float* __restrict__ psq)
{
    __shared__ float sRowsF[ROWS_PER_BLK * HH];    // 32 KB, e tile in f32
    __shared__ float sVi[HH];
    __shared__ float sB[HH];
    __shared__ float sRedS[128 * 8];               // 4 KB
    __shared__ float sRedQ[128 * 8];               // 4 KB

    const int t  = threadIdx.x;
    const int bi = blockIdx.x;
    const int r0 = bi * ROWS_PER_BLK;
    const int i_node = bi / 12;                    // 768/64 = 12 tiles per i
    const int j0 = (bi % 12) * ROWS_PER_BLK;

    // ---- async stage: 64 rows (32 KB) global -> LDS, 16 B per lane per op
    {
        const char* gbase = (const char*)(e + (size_t)r0 * HH);
        unsigned lbase = (unsigned)(size_t)(&sRowsF[0]);   // LDS byte offset (low 32b of aperture addr)
#pragma unroll
        for (int q = 0; q < 16; ++q) {
            unsigned byteoff = (unsigned)(t + q * 128) * 16u;
            unsigned laddr = lbase + byteoff;
            unsigned long long gaddr = (unsigned long long)(gbase + byteoff);
            asm volatile("global_load_async_to_lds_b128 %0, %1, off"
                         :: "v"(laddr), "v"(gaddr) : "memory");
        }
    }
    if (t < HH) { sVi[t] = Vx[(size_t)i_node * HH + t]; sB[t] = bias[t]; }
    asm volatile("s_wait_asynccnt 0x0" ::: "memory");
    __syncthreads();

    const int wave  = t >> 5;       // 0..3
    const int lane  = t & 31;
    const int lhalf = lane >> 4;    // 0 or 1
    const int l15   = lane & 15;
    const int mbase = wave * 16;    // each wave: 16 rows

    v8f acc[8] = {};                // 8 channel tiles of 16x16 f32

    const int arow  = mbase + l15;
    const int akoff = lhalf * 16;
#pragma unroll
    for (int kb = 0; kb < HH; kb += 32) {
        // A fragment (16x32 f16): lane<16 holds K {0..15}, lane>=16 {16..31}
        const float* pa = &sRowsF[arow * HH + kb + akoff];
        float4 a0 = *(const float4*)(pa);
        float4 a1 = *(const float4*)(pa + 4);
        float4 a2 = *(const float4*)(pa + 8);
        float4 a3 = *(const float4*)(pa + 12);
        v16h afrag = { (_Float16)a0.x, (_Float16)a0.y, (_Float16)a0.z, (_Float16)a0.w,
                       (_Float16)a1.x, (_Float16)a1.y, (_Float16)a1.z, (_Float16)a1.w,
                       (_Float16)a2.x, (_Float16)a2.y, (_Float16)a2.z, (_Float16)a2.w,
                       (_Float16)a3.x, (_Float16)a3.y, (_Float16)a3.z, (_Float16)a3.w };
#pragma unroll
        for (int ct = 0; ct < 8; ++ct) {
            // B fragment (32x16 f16): column g = ct*16+l15, 16 contiguous K halves
            int g = ct * 16 + l15;
            const _Float16* pb = Wh + (size_t)g * HH + kb + lhalf * 16;
            v16h bfrag = *(const v16h*)pb;
            acc[ct] = __builtin_amdgcn_wmma_f32_16x16x32_f16(
                false, afrag, false, bfrag, (short)0, acc[ct], false, false);
        }
    }

    // epilogue: bias, (Vx[i]+Vx[j]) gate, store e_new, BN partial sums
#pragma unroll
    for (int ct = 0; ct < 8; ++ct) {
        int g = ct * 16 + l15;
        float vxi = sVi[g];
        float bb  = sB[g];
        float ls = 0.f, lq = 0.f;
#pragma unroll
        for (int v = 0; v < 8; ++v) {
            int mrow = mbase + v + lhalf * 8;      // C layout: lanes>=16 -> M+8
            int j = j0 + mrow;
            float vs  = vxi + Vx[(size_t)j * HH + g];
            float val = (acc[ct][v] + bb) * vs;
            enew[((size_t)(r0 + mrow)) * HH + g] = val;
            ls += val; lq = fmaf(val, val, lq);
        }
        sRedS[t * 8 + ct] = ls;
        sRedQ[t * 8 + ct] = lq;
    }
    __syncthreads();
    // deterministic per-channel reduction (fixed order)
    if (t < HH) {
        int ct = t >> 4, gl = t & 15;
        float s = 0.f, q = 0.f;
#pragma unroll
        for (int w = 0; w < 4; ++w) {
            int l0 = w * 32 + gl, l1 = l0 + 16;
            s += sRedS[l0 * 8 + ct] + sRedS[l1 * 8 + ct];
            q += sRedQ[l0 * 8 + ct] + sRedQ[l1 * 8 + ct];
        }
        psum[(size_t)bi * HH + t] = s;
        psq [(size_t)bi * HH + t] = q;
    }
}

// ---------------- BN reduce over blocks + build affine --------------------
__global__ void bn_reduce_kernel(const float* __restrict__ psum, const float* __restrict__ psq,
                                 const float* __restrict__ gamma, const float* __restrict__ beta,
                                 float* __restrict__ scale, float* __restrict__ shift)
{
    __shared__ float ss[256], sq[256];
    int ch = blockIdx.x, t = threadIdx.x;
    float s = 0.f, q = 0.f;
    for (int p = t; p < GEMM_BLOCKS; p += 256) {
        s += psum[(size_t)p * HH + ch];
        q += psq [(size_t)p * HH + ch];
    }
    ss[t] = s; sq[t] = q; __syncthreads();
    for (int off = 128; off > 0; off >>= 1) {
        if (t < off) { ss[t] += ss[t + off]; sq[t] += sq[t + off]; }
        __syncthreads();
    }
    if (t == 0) {
        float inv  = 1.0f / (float)NROWS;
        float mean = ss[0] * inv;
        float var  = sq[0] * inv - mean * mean;
        float rstd = rsqrtf(var + 1e-5f);
        float sc   = gamma[ch] * rstd;
        scale[ch] = sc;
        shift[ch] = beta[ch] - mean * sc;
    }
}

// ---------------- online softmax stats over neighbor axis -----------------
__global__ void softmax_stats_kernel(const float* __restrict__ enew,
                                     const float* __restrict__ scale, const float* __restrict__ shift,
                                     const float* __restrict__ Vx2,
                                     float* __restrict__ M, float* __restrict__ S, float* __restrict__ Wn)
{
    int i = blockIdx.x, h = threadIdx.x;           // 128 threads, coalesced on h
    float sc = scale[h], sh = shift[h];
    const float* row = enew + (size_t)i * NN * HH + h;
    float m = -FLT_MAX, s = 0.f, w = 0.f;
    for (int j = 0; j < NN; ++j) {
        float z  = fmaf(sc, row[(size_t)j * HH], sh);
        float mn = fmaxf(m, z);
        float c  = __expf(m - mn);
        float ez = __expf(z - mn);
        s = fmaf(s, c, ez);
        w = fmaf(w, c, ez * Vx2[(size_t)j * HH + h]);
        m = mn;
    }
    M [(size_t)i * HH + h] = m;
    S [(size_t)i * HH + h] = s;
    Wn[(size_t)i * HH + h] = w;
}

// ---------------- node update: BN over nodes + relu + residual ------------
__global__ void node_update_kernel(const float* __restrict__ x, const float* __restrict__ Ux,
                                   const float* __restrict__ S, const float* __restrict__ Wn,
                                   const float* __restrict__ gamma, const float* __restrict__ beta,
                                   float* __restrict__ xout)
{
    __shared__ float ss[256], sq[256];
    __shared__ float sstat[2];
    int h = blockIdx.x, t = threadIdx.x;
    float xv[3];
    float s = 0.f, q = 0.f;
#pragma unroll
    for (int k = 0; k < 3; ++k) {                  // 768 = 3*256 nodes
        int i = t + k * 256;
        size_t id = (size_t)i * HH + h;
        float num = Wn[id] / S[id];
        float v = Ux[id] + num / (1.0f + 1e-20f);  // den = eps + sum(softmax) = 1+eps
        xv[k] = v; s += v; q = fmaf(v, v, q);
    }
    ss[t] = s; sq[t] = q; __syncthreads();
    for (int off = 128; off > 0; off >>= 1) {
        if (t < off) { ss[t] += ss[t + off]; sq[t] += sq[t + off]; }
        __syncthreads();
    }
    if (t == 0) {
        float inv  = 1.0f / 768.0f;
        float mean = ss[0] * inv;
        float var  = sq[0] * inv - mean * mean;
        float rstd = rsqrtf(var + 1e-5f);
        float sc   = gamma[h] * rstd;
        sstat[0] = sc; sstat[1] = beta[h] - mean * sc;
    }
    __syncthreads();
    float sc = sstat[0], sh = sstat[1];
#pragma unroll
    for (int k = 0; k < 3; ++k) {
        int i = t + k * 256;
        size_t id = (size_t)i * HH + h;
        float bn = fmaf(sc, xv[k], sh);
        xout[id] = x[id] + fmaxf(bn, 0.f);
    }
}

// ---------------- e_next = e * (1 + gate), in place over e_new ------------
__global__ void edge_out_kernel(float* __restrict__ eio, const float* __restrict__ e,
                                const float* __restrict__ scale, const float* __restrict__ shift,
                                const float* __restrict__ M, const float* __restrict__ S)
{
    size_t idx4 = (size_t)blockIdx.x * 256 + threadIdx.x;
    size_t base = idx4 * 4;
    int h = (int)(base & 127);
    size_t row = base >> 7;
    int i = (int)(row / NN);
    float4 zn = *(const float4*)(eio + base);
    float4 ev = *(const float4*)(e + base);
    float4 sc = *(const float4*)(scale + h);
    float4 sh = *(const float4*)(shift + h);
    float4 mm = *(const float4*)(M + (size_t)i * HH + h);
    float4 sv = *(const float4*)(S + (size_t)i * HH + h);
    float4 o;
    o.x = ev.x * (1.f + __expf(fmaf(sc.x, zn.x, sh.x) - mm.x) / sv.x);
    o.y = ev.y * (1.f + __expf(fmaf(sc.y, zn.y, sh.y) - mm.y) / sv.y);
    o.z = ev.z * (1.f + __expf(fmaf(sc.z, zn.z, sh.z) - mm.z) / sv.z);
    o.w = ev.w * (1.f + __expf(fmaf(sc.w, zn.w, sh.w) - mm.w) / sv.w);
    *(float4*)(eio + base) = o;
}

extern "C" void kernel_launch(void* const* d_in, const int* in_sizes, int n_in,
                              void* d_out, int out_size, void* d_ws, size_t ws_size,
                              hipStream_t stream) {
    (void)in_sizes; (void)n_in; (void)out_size; (void)ws_size;
    const float* x      = (const float*)d_in[0];
    const float* e      = (const float*)d_in[1];
    const float* eU_w   = (const float*)d_in[2];
    const float* eU_b   = (const float*)d_in[3];
    const float* eV_w   = (const float*)d_in[4];
    const float* eV_b   = (const float*)d_in[5];
    const float* nU_w   = (const float*)d_in[6];
    const float* nU_b   = (const float*)d_in[7];
    const float* nV_w   = (const float*)d_in[8];
    const float* nV_b   = (const float*)d_in[9];
    const float* e_gamma = (const float*)d_in[10];
    const float* e_beta  = (const float*)d_in[11];
    const float* n_gamma = (const float*)d_in[12];
    const float* n_beta  = (const float*)d_in[13];

    float* ws   = (float*)d_ws;                 // ~11.8 MB used
    float* Vx   = ws;                           // 98304
    float* Vx2  = ws + 98304;                   // 98304
    float* Ux   = ws + 196608;                  // 98304
    float* M    = ws + 294912;                  // 98304
    float* S    = ws + 393216;                  // 98304
    float* Wn   = ws + 491520;                  // 98304
    float* scale = ws + 589824;                 // 128
    float* shift = ws + 589952;                 // 128
    float* psum = ws + 590080;                  // 9216*128
    float* psq  = ws + 1769728;                 // 9216*128
    _Float16* Wh = (_Float16*)(ws + 2949376);   // 16384 halves (32-byte aligned)

    float* xout = (float*)d_out;                // [768,128]
    float* eout = (float*)d_out + 98304;        // [768,768,128], e_new then e_next in place

    linear_kernel<<<384, 256, 0, stream>>>(x, eV_w, eV_b, Vx);
    linear_kernel<<<384, 256, 0, stream>>>(x, nU_w, nU_b, Ux);
    linear_kernel<<<384, 256, 0, stream>>>(x, nV_w, nV_b, Vx2);
    cvt_f16_kernel<<<64, 256, 0, stream>>>(eU_w, Wh);

    edge_gemm_kernel<<<GEMM_BLOCKS, 128, 0, stream>>>(e, Wh, eU_b, Vx, eout, psum, psq);
    bn_reduce_kernel<<<128, 256, 0, stream>>>(psum, psq, e_gamma, e_beta, scale, shift);
    softmax_stats_kernel<<<768, 128, 0, stream>>>(eout, scale, shift, Vx2, M, S, Wn);
    node_update_kernel<<<128, 256, 0, stream>>>(x, Ux, S, Wn, n_gamma, n_beta, xout);
    edge_out_kernel<<<73728, 256, 0, stream>>>(eout, e, scale, shift, M, S);
}
